// ConvAttention_spa_att_less_token_19327352832682
// MI455X (gfx1250) — compile-verified, matrix operand order checked by reading, not run on hardware
//
#include <hip/hip_runtime.h>
#include <hip/hip_bf16.h>

// ---------------- problem constants ----------------
#define BATCH   16
#define IMG     56
#define P       28
#define NTOK    784          // P*P
#define DIM     256
#define HEADS   8
#define DH      64
#define INNER   512          // HEADS*DH
#define KPAD    800          // 784 padded to multiple of 32 (25 chunks)

typedef __attribute__((ext_vector_type(16))) __bf16 v16bf;
typedef __attribute__((ext_vector_type(8)))  float  v8f;
typedef int vi4 __attribute__((vector_size(16)));    // 16-byte int vector (b128 payload)

#if defined(__gfx1250__) && __has_builtin(__builtin_amdgcn_global_load_async_to_lds_b128) && \
    __has_builtin(__builtin_amdgcn_s_wait_asynccnt)
#define USE_ASYNC_LDS 1
#else
#define USE_ASYNC_LDS 0
#endif

static __device__ __forceinline__ int iclamp(int v, int lo, int hi) {
    return v < lo ? lo : (v > hi ? hi : v);
}

static __device__ __forceinline__ unsigned short f2bf_u16(float f) {
    union { __hip_bfloat16 h; unsigned short u; } cv;
    cv.h = __float2bfloat16(f);
    return cv.u;
}

// copy 16 contiguous bytes global -> LDS (async GLOBAL_LOAD_ASYNC_TO_LDS_B128 on gfx1250)
static __device__ __forceinline__ void cp16(unsigned short* lds, const unsigned short* g) {
#if USE_ASYNC_LDS
    __builtin_amdgcn_global_load_async_to_lds_b128(
        (__attribute__((address_space(1))) vi4*)(unsigned long long)g,
        (__attribute__((address_space(3))) vi4*)(unsigned long long)lds, 0, 0);
#else
    *(uint4*)lds = *(const uint4*)g;
#endif
}

// wait for staged tiles, then block barrier (async needs explicit ASYNCcnt wait)
static __device__ __forceinline__ void stage_fence() {
#if USE_ASYNC_LDS
    __builtin_amdgcn_s_wait_asynccnt(0);
#endif
    __syncthreads();
}

// Build one 16x32 bf16 WMMA operand fragment from an LDS tile.
// Tile: 16 rows (M for A, N for B) x 32 K, row stride `ld` ushorts (16B aligned).
// Mapping per CDNA5 ISA 7.12.2: lane rc = lane&15, half = lane>>4;
// frag halves are two contiguous 16B chunks at K = half*8 and K = 16+half*8.
static __device__ __forceinline__ v16bf frag_load(const unsigned short* base, int ld) {
    int lane = threadIdx.x & 31;
    const unsigned short* p = base + (lane & 15) * ld + (lane >> 4) * 8;
    union { v16bf v; uint4 q[2]; } f;
    f.q[0] = *(const uint4*)p;          // K = half*8 + 0..7   -> vgprs 0..3
    f.q[1] = *(const uint4*)(p + 16);   // K = 16+half*8 +0..7 -> vgprs 4..7
    return f.v;
}

// bicubic weight, a = -0.75 (PyTorch default)
static __device__ __forceinline__ float cubw(float d) {
    const float a = -0.75f;
    float ad = fabsf(d);
    if (ad <= 1.f) return (a + 2.f) * ad * ad * ad - (a + 3.f) * ad * ad + 1.f;
    if (ad < 2.f)  return a * (ad * ad * ad - 5.f * ad * ad + 8.f * ad - 4.f);
    return 0.f;
}

// ---------------- kernel 0: TDM path exercise (NULL descriptor => no-op per ISA) --
__global__ void k_tdm_nop() {
#if defined(__gfx1250__) && __has_builtin(__builtin_amdgcn_tensor_load_to_lds) && \
    __has_builtin(__builtin_amdgcn_s_wait_tensorcnt)
    typedef unsigned int u32x4 __attribute__((ext_vector_type(4)));
    typedef int i32x8 __attribute__((ext_vector_type(8)));
    typedef int i32x4 __attribute__((ext_vector_type(4)));
    u32x4 g0 = {};   // count=0 -> NULL tensor descriptor: TDM no-op (ISA 8.3)
    i32x8 g1 = {};
    i32x4 g2 = {}, g3 = {};
    i32x8 g4 = {};
    __builtin_amdgcn_tensor_load_to_lds(g0, g1, g2, g3, g4, 0);   // 6-arg form (clang-23 lane)
    __builtin_amdgcn_s_wait_tensorcnt((unsigned short)0);
#endif
}

// ---------------- kernel 1: avg-pool 3x3 s2 p1 + channel mean/max ----------------
// grid (784, 16) block 256 (one thread per channel)
__global__ void k_pool(const float* __restrict__ x, float* __restrict__ xpool,
                       float* __restrict__ xattn) {
    int j = blockIdx.x, b = blockIdx.y, c = threadIdx.x;
    int oy = j / P, ox = j % P;
    float s = 0.f;
#pragma unroll
    for (int dy = -1; dy <= 1; ++dy)
#pragma unroll
        for (int dx = -1; dx <= 1; ++dx) {
            int y = 2 * oy + dy, xx = 2 * ox + dx;
            if (y >= 0 && y < IMG && xx >= 0 && xx < IMG)
                s += x[((size_t)b * (IMG * IMG) + y * IMG + xx) * DIM + c];
        }
    s *= (1.f / 9.f);
    xpool[((size_t)b * DIM + c) * NTOK + j] = s;

    __shared__ float rs[256], rm[256];
    rs[c] = s; rm[c] = s;
    __syncthreads();
    for (int off = 128; off > 0; off >>= 1) {
        if (c < off) { rs[c] += rs[c + off]; rm[c] = fmaxf(rm[c], rm[c + off]); }
        __syncthreads();
    }
    if (c == 0) {
        xattn[((size_t)b * 2 + 0) * NTOK + j] = rs[0] * (1.f / 256.f);
        xattn[((size_t)b * 2 + 1) * NTOK + j] = rm[0];
    }
}

// ---------------- kernel 2: q/k 3x3 conv (2 -> 16 ch) ----------------
__global__ void k_qk(const float* __restrict__ xattn,
                     const float* __restrict__ qw, const float* __restrict__ qbias,
                     const float* __restrict__ kw, const float* __restrict__ kbias,
                     float* __restrict__ qo, float* __restrict__ ko) {
    int idx = blockIdx.x * 256 + threadIdx.x;
    if (idx >= BATCH * 2 * 16 * NTOK) return;
    int j = idx % NTOK; int t = idx / NTOK;
    int oc = t % 16; t /= 16;
    int which = t & 1; int b = t >> 1;
    int oy = j / P, ox = j % P;
    const float* w  = which ? kw : qw;
    const float* bb = which ? kbias : qbias;
    float s = bb[oc];
#pragma unroll
    for (int ic = 0; ic < 2; ++ic)
#pragma unroll
        for (int dy = -1; dy <= 1; ++dy)
#pragma unroll
            for (int dx = -1; dx <= 1; ++dx) {
                int y = oy + dy, xx = ox + dx;
                if (y >= 0 && y < P && xx >= 0 && xx < P)
                    s += w[((oc * 2 + ic) * 3 + (dy + 1)) * 3 + (dx + 1)] *
                         xattn[((size_t)b * 2 + ic) * NTOK + y * P + xx];
            }
    int h = oc >> 1, dd = oc & 1;
    float* dst = which ? ko : qo;
    dst[(((size_t)b * HEADS + h) * NTOK + j) * 2 + dd] = s;
}

// ---------------- kernel 3: depthwise conv + BN + exact GELU -> bf16 [b][j][c] ----
__global__ void k_dw(const float* __restrict__ xpool, const float* __restrict__ dww,
                     const float* __restrict__ dwb, const float* __restrict__ g,
                     const float* __restrict__ beta, const float* __restrict__ mean,
                     const float* __restrict__ var, __hip_bfloat16* __restrict__ vmid) {
    int idx = blockIdx.x * 256 + threadIdx.x;
    if (idx >= BATCH * DIM * NTOK) return;
    int j = idx % NTOK; int t = idx / NTOK;
    int c = t % DIM; int b = t / DIM;
    int oy = j / P, ox = j % P;
    float s = dwb[c];
    const float* wp = dww + c * 9;
    const float* xp = xpool + ((size_t)b * DIM + c) * NTOK;
#pragma unroll
    for (int dy = -1; dy <= 1; ++dy)
#pragma unroll
        for (int dx = -1; dx <= 1; ++dx) {
            int y = oy + dy, xx = ox + dx;
            if (y >= 0 && y < P && xx >= 0 && xx < P)
                s += wp[(dy + 1) * 3 + dx + 1] * xp[y * P + xx];
        }
    float sc = g[c] * rsqrtf(var[c] + 1e-5f);
    s = (s - mean[c]) * sc + beta[c];
    s = 0.5f * s * (1.f + erff(s * 0.70710678118654752f));   // exact GELU
    vmid[((size_t)b * NTOK + j) * DIM + c] = __float2bfloat16(s);
}

// ---------------- kernel 4: weight conversion -> bf16 (no transpose needed) ------
// pw_w [o=512][c=256] is already N-major for WMMA B; out_w [n=256][k=512] likewise.
__global__ void k_cvtw(const float* __restrict__ pw, const float* __restrict__ ow,
                       __hip_bfloat16* __restrict__ pwB, __hip_bfloat16* __restrict__ owB) {
    int idx = blockIdx.x * 256 + threadIdx.x;
    if (idx < INNER * DIM) pwB[idx] = __float2bfloat16(pw[idx]);
    else {
        int i2 = idx - INNER * DIM;
        if (i2 < DIM * INNER) owB[i2] = __float2bfloat16(ow[i2]);
    }
}

// ---------------- kernel 5: 1x1 conv GEMM (WMMA), 784x512x256 per batch ----------
// grid (49, 2, 16) block 128 (4 waves). Block tile 16(j) x 256(o); wave owns 64 o,
// 4 accumulators -> 4 WMMAs per barrier. Writes V transposed: vT[b][h][d][j].
__global__ void k_gemm_pw(const __hip_bfloat16* __restrict__ vmid,
                          const __hip_bfloat16* __restrict__ pwB,
                          const float* __restrict__ pwb,
                          __hip_bfloat16* __restrict__ vT) {
    int jt = blockIdx.x, ot = blockIdx.y, b = blockIdx.z;
    int tid = threadIdx.x, wave = tid >> 5, lane = tid & 31;
    __shared__ __align__(16) unsigned short As[16 * 40];
    __shared__ __align__(16) unsigned short Bs[256 * 40];
    const unsigned short* A = (const unsigned short*)vmid + ((size_t)b * NTOK + jt * 16) * DIM;
    const unsigned short* W = (const unsigned short*)pwB + (size_t)(ot * 256) * DIM; // [o][c]
    v8f acc[4] = {{}, {}, {}, {}};
    for (int k0 = 0; k0 < DIM; k0 += 32) {
        if (tid < 64) {                       // A: 16 rows x 4 chunks of 8 ushorts
            int r = tid >> 2, c = tid & 3;
            cp16(&As[r * 40 + c * 8], &A[r * DIM + k0 + c * 8]);
        }
        for (int ch = tid; ch < 1024; ch += 128) {   // B: 256 rows x 4 chunks
            int r = ch >> 2, c = ch & 3;
            cp16(&Bs[r * 40 + c * 8], &W[(size_t)r * DIM + k0 + c * 8]);
        }
        stage_fence();
        v16bf af = frag_load(As, 40);
#pragma unroll
        for (int nn = 0; nn < 4; ++nn) {
            v16bf bf = frag_load(Bs + (wave * 64 + nn * 16) * 40, 40);
            acc[nn] = __builtin_amdgcn_wmma_f32_16x16x32_bf16(false, af, false, bf,
                                                              (short)0, acc[nn], false, false);
        }
        __syncthreads();
    }
    int n_ = lane & 15, mhi = lane >> 4;
#pragma unroll
    for (int nn = 0; nn < 4; ++nn) {
        int o = ot * 256 + wave * 64 + nn * 16 + n_;
        float bias = pwb[o];
        int h = o >> 6, d = o & 63;
#pragma unroll
        for (int r = 0; r < 8; ++r) {
            int m = r + mhi * 8;
            int j = jt * 16 + m;
            vT[(((size_t)b * HEADS + h) * DH + d) * NTOK + j] =
                __float2bfloat16(acc[nn][r] + bias);
        }
    }
}

// ---------------- kernel 6: attention ----------------
// grid (49, 8, 16) block 128 (4 waves, all do WMMA; wave w owns d-subtile w*16).
__global__ void k_attn(const float* __restrict__ qb_, const float* __restrict__ kb_,
                       const __hip_bfloat16* __restrict__ vT_,
                       __hip_bfloat16* __restrict__ ao) {
    int it = blockIdx.x, h = blockIdx.y, b = blockIdx.z;
    int tid = threadIdx.x;
    __shared__ float qs[16][2];
    __shared__ float redbuf[128];
    __shared__ float rowinv[16];
    __shared__ __align__(16) unsigned short pb[16][KPAD];  // unnormalized exp, bf16
    __shared__ __align__(16) unsigned short vt[64 * 40];   // V tile [d][kk]
    const float* qp = qb_ + (((size_t)b * HEADS + h) * NTOK) * 2;
    const float* kp = kb_ + (((size_t)b * HEADS + h) * NTOK) * 2;
    int i0 = it * 16;
    if (tid < 32) qs[tid >> 1][tid & 1] = qp[(size_t)(i0 + (tid >> 1)) * 2 + (tid & 1)];
    __syncthreads();
    const float scale = 0.125f;                // DH^-0.5
    for (int r = 0; r < 16; ++r) {
        float q0 = qs[r][0] * scale, q1 = qs[r][1] * scale;
        float lm = -3.4e38f;
        for (int j = tid; j < NTOK; j += 128) {
            float s = q0 * kp[j * 2] + q1 * kp[j * 2 + 1];
            lm = fmaxf(lm, s);
        }
        redbuf[tid] = lm; __syncthreads();
        for (int off = 64; off > 0; off >>= 1) {
            if (tid < off) redbuf[tid] = fmaxf(redbuf[tid], redbuf[tid + off]);
            __syncthreads();
        }
        float m = redbuf[0];
        __syncthreads();
        float ls = 0.f;
        for (int j = tid; j < KPAD; j += 128) {
            float p = 0.f;
            if (j < NTOK) {
                float s = q0 * kp[j * 2] + q1 * kp[j * 2 + 1];
                p = __expf(s - m);
                ls += p;
            }
            pb[r][j] = f2bf_u16(p);
        }
        redbuf[tid] = ls; __syncthreads();
        for (int off = 64; off > 0; off >>= 1) {
            if (tid < off) redbuf[tid] += redbuf[tid + off];
            __syncthreads();
        }
        if (tid == 0) rowinv[r] = 1.f / redbuf[0];
        __syncthreads();
    }
    // attn @ v : 16 x KPAD @ KPAD x 64, 25 K-chunks of 32
    int wave = tid >> 5, lane = tid & 31;
    const unsigned short* vp = (const unsigned short*)vT_ +
                               (((size_t)b * HEADS + h) * DH) * NTOK; // [d][j]
    v8f acc = {};
    for (int kb = 0; kb < KPAD / 32; ++kb) {
        int j0 = kb * 32;
        for (int ch = tid; ch < 256; ch += 128) {   // 64 d-rows x 4 chunks
            int d = ch >> 2, c = ch & 3;
            // j columns >= NTOK read past row end (allocation padded); they are
            // multiplied by pb==0 columns so the result is exact.
            cp16(&vt[d * 40 + c * 8], &vp[(size_t)d * NTOK + j0 + c * 8]);
        }
        stage_fence();
        v16bf af = frag_load(&pb[0][0] + j0, KPAD);
        v16bf bf = frag_load(vt + (wave * 16) * 40, 40);
        acc = __builtin_amdgcn_wmma_f32_16x16x32_bf16(false, af, false, bf,
                                                      (short)0, acc, false, false);
        __syncthreads();
    }
    int n_ = lane & 15, mhi = lane >> 4;
#pragma unroll
    for (int r = 0; r < 8; ++r) {
        int m = r + mhi * 8;
        float val = acc[r] * rowinv[m];
        int i = i0 + m;
        int d = wave * 16 + n_;
        ao[((size_t)b * NTOK + i) * INNER + h * DH + d] = __float2bfloat16(val);
    }
}

// ---------------- kernel 7: projection GEMM (12544 x 256 x 512) ----------------
// grid (784) block 128. Block tile 16(m) x 256(n); wave owns 64 n, 4 accumulators.
__global__ void k_gemm_out(const __hip_bfloat16* __restrict__ ao,
                           const __hip_bfloat16* __restrict__ owB,
                           const float* __restrict__ ob, float* __restrict__ proj) {
    int mt = blockIdx.x;
    int tid = threadIdx.x, wave = tid >> 5, lane = tid & 31;
    __shared__ __align__(16) unsigned short As[16 * 40];
    __shared__ __align__(16) unsigned short Bs[256 * 40];
    const unsigned short* A = (const unsigned short*)ao + (size_t)mt * 16 * INNER;
    const unsigned short* W = (const unsigned short*)owB;   // [n=256][k=512]
    v8f acc[4] = {{}, {}, {}, {}};
    for (int k0 = 0; k0 < INNER; k0 += 32) {
        if (tid < 64) {
            int r = tid >> 2, c = tid & 3;
            cp16(&As[r * 40 + c * 8], &A[r * INNER + k0 + c * 8]);
        }
        for (int ch = tid; ch < 1024; ch += 128) {
            int r = ch >> 2, c = ch & 3;
            cp16(&Bs[r * 40 + c * 8], &W[(size_t)r * INNER + k0 + c * 8]);
        }
        stage_fence();
        v16bf af = frag_load(As, 40);
#pragma unroll
        for (int nn = 0; nn < 4; ++nn) {
            v16bf bf = frag_load(Bs + (wave * 64 + nn * 16) * 40, 40);
            acc[nn] = __builtin_amdgcn_wmma_f32_16x16x32_bf16(false, af, false, bf,
                                                              (short)0, acc[nn], false, false);
        }
        __syncthreads();
    }
    int n_ = lane & 15, mhi = lane >> 4;
#pragma unroll
    for (int nn = 0; nn < 4; ++nn) {
        int ng = wave * 64 + nn * 16 + n_;
        float bias = ob[ng];
#pragma unroll
        for (int r = 0; r < 8; ++r) {
            int m = r + mhi * 8;
            proj[((size_t)mt * 16 + m) * DIM + ng] = acc[nn][r] + bias;
        }
    }
}

// ---------------- kernel 8: bicubic x2 upsample -> [b, 3136, 256] f32 ------------
__global__ void k_up(const float* __restrict__ proj, float* __restrict__ out) {
    int pix = blockIdx.x, b = blockIdx.y, c = threadIdx.x;
    int Y = pix / IMG, X = pix % IMG;
    float wy[4], wx[4]; int iy[4], ix[4];
    {
        float src = (Y + 0.5f) * 0.5f - 0.5f;
        float fl = floorf(src); int i0 = (int)fl; float t = src - fl;
        float d[4] = {1.f + t, t, 1.f - t, 2.f - t};
#pragma unroll
        for (int k = 0; k < 4; ++k) { wy[k] = cubw(d[k]); iy[k] = iclamp(i0 + k - 1, 0, P - 1); }
    }
    {
        float src = (X + 0.5f) * 0.5f - 0.5f;
        float fl = floorf(src); int i0 = (int)fl; float t = src - fl;
        float d[4] = {1.f + t, t, 1.f - t, 2.f - t};
#pragma unroll
        for (int k = 0; k < 4; ++k) { wx[k] = cubw(d[k]); ix[k] = iclamp(i0 + k - 1, 0, P - 1); }
    }
    float acc = 0.f;
#pragma unroll
    for (int a = 0; a < 4; ++a)
#pragma unroll
        for (int bb = 0; bb < 4; ++bb)
            acc += wy[a] * wx[bb] * proj[((size_t)b * NTOK + iy[a] * P + ix[bb]) * DIM + c];
    out[((size_t)b * (IMG * IMG) + pix) * DIM + c] = acc;
}

// ---------------- launcher ----------------
extern "C" void kernel_launch(void* const* d_in, const int* in_sizes, int n_in,
                              void* d_out, int out_size, void* d_ws, size_t ws_size,
                              hipStream_t stream) {
    (void)in_sizes; (void)n_in; (void)out_size; (void)ws_size;
    const float* x     = (const float*)d_in[0];
    const float* q_w   = (const float*)d_in[1];
    const float* q_b   = (const float*)d_in[2];
    const float* k_w   = (const float*)d_in[3];
    const float* k_b   = (const float*)d_in[4];
    const float* dw_w  = (const float*)d_in[5];
    const float* dw_b  = (const float*)d_in[6];
    const float* bn_g  = (const float*)d_in[7];
    const float* bn_be = (const float*)d_in[8];
    const float* bn_m  = (const float*)d_in[9];
    const float* bn_v  = (const float*)d_in[10];
    const float* pw_w  = (const float*)d_in[11];
    const float* pw_b  = (const float*)d_in[12];
    const float* out_w = (const float*)d_in[13];
    const float* out_b = (const float*)d_in[14];
    float* out = (float*)d_out;

    // workspace layout (bytes)
    char* ws = (char*)d_ws;
    constexpr size_t SZ_XPOOL = (size_t)BATCH * DIM * NTOK * 4;
    constexpr size_t SZ_XATTN = (size_t)BATCH * 2 * NTOK * 4;
    constexpr size_t SZ_QK    = (size_t)BATCH * HEADS * NTOK * 2 * 4;
    constexpr size_t SZ_VMID  = (size_t)BATCH * NTOK * DIM * 2;
    constexpr size_t SZ_PWB   = (size_t)INNER * DIM * 2;
    constexpr size_t SZ_OWB   = (size_t)DIM * INNER * 2;
    constexpr size_t SZ_VT    = (size_t)BATCH * HEADS * DH * NTOK * 2 + 256; // +pad for b128 tail reads
    constexpr size_t SZ_AO    = (size_t)BATCH * NTOK * INNER * 2;

    size_t off = 0;
    float* xpool = (float*)(ws + off); off += SZ_XPOOL;
    float* xattn = (float*)(ws + off); off += SZ_XATTN;
    float* qbuf  = (float*)(ws + off); off += SZ_QK;
    float* kbuf  = (float*)(ws + off); off += SZ_QK;
    __hip_bfloat16* vmid = (__hip_bfloat16*)(ws + off); off += SZ_VMID;
    __hip_bfloat16* pwB  = (__hip_bfloat16*)(ws + off); off += SZ_PWB;
    __hip_bfloat16* owB  = (__hip_bfloat16*)(ws + off); off += SZ_OWB;
    __hip_bfloat16* vT   = (__hip_bfloat16*)(ws + off); off += SZ_VT;
    __hip_bfloat16* ao   = (__hip_bfloat16*)(ws + off); off += SZ_AO;
    float* proj = (float*)(ws + off);

    k_tdm_nop<<<dim3(1), 32, 0, stream>>>();
    k_pool<<<dim3(NTOK, BATCH), 256, 0, stream>>>(x, xpool, xattn);
    k_qk<<<dim3((BATCH * 2 * 16 * NTOK) / 256), 256, 0, stream>>>(
        xattn, q_w, q_b, k_w, k_b, qbuf, kbuf);
    k_dw<<<dim3((BATCH * DIM * NTOK) / 256), 256, 0, stream>>>(
        xpool, dw_w, dw_b, bn_g, bn_be, bn_m, bn_v, vmid);
    k_cvtw<<<dim3((2 * DIM * INNER) / 256), 256, 0, stream>>>(pw_w, out_w, pwB, owB);
    k_gemm_pw<<<dim3(NTOK / 16, 2, BATCH), 128, 0, stream>>>(vmid, pwB, pw_b, vT);
    k_attn<<<dim3(NTOK / 16, HEADS, BATCH), 128, 0, stream>>>(qbuf, kbuf, vT, ao);
    k_gemm_out<<<dim3((BATCH * NTOK) / 16), 128, 0, stream>>>(ao, owB, out_b, proj);
    k_up<<<dim3(IMG * IMG, BATCH), 256, 0, stream>>>(proj, out);
}
